// TTAggregator_84293028151830
// MI455X (gfx1250) — matrix operand order repaired and checked by reading, not compiled, not run on hardware
//
#include <hip/hip_runtime.h>

// TT aggregation: out[n] = (Prod_{d=7..1} reshape(h_d @ U_list[d], 8,8)) @ (h_0 @ U_first) @ U_last
// GEMM-dominant part mapped to v_wmma_f32_16x16x32_f16 (f16 in, f32 accumulate).
// All WMMA B-operands are pre-staged to LDS in fragment layout (incl. zero padding)
// so the hot path has no divergent scalar gathers.

namespace {
constexpr int kN     = 150000;
constexpr int kD     = 8;
constexpr int kIn    = 256;
constexpr int kR     = 8;
constexpr int kRR    = 64;
constexpr int kOut   = 256;
constexpr int kWaves = 12;                 // 384 threads / block
constexpr int kTiles = (kN + 15) / 16;     // 9375 sample tiles of 16
}

typedef __attribute__((ext_vector_type(16))) _Float16 v16h;
typedef __attribute__((ext_vector_type(8)))  float    v8f;

#define WMMA_F32_F16(acc, a, b)                                              \
  (acc) = __builtin_amdgcn_wmma_f32_16x16x32_f16(false, (a), false, (b),     \
                                                 (short)0, (acc), false, false)

// A-fragment (16x32 f16): lane m = lane&15 holds row m, lane group g = lane>>4
// selects K sub-blocks. Element e: K = g*16 + e.
__device__ __forceinline__ v16h make_a_frag(const float* __restrict__ hp, int kb, int g) {
  const int kA = kb * 32 + g * 16;
  const float4 f0 = *(const float4*)(hp + kA);
  const float4 f1 = *(const float4*)(hp + kA + 4);
  const float4 f2 = *(const float4*)(hp + kA + 8);
  const float4 f3 = *(const float4*)(hp + kA + 12);
  v16h a;
  a[0]  = (_Float16)f0.x; a[1]  = (_Float16)f0.y;
  a[2]  = (_Float16)f0.z; a[3]  = (_Float16)f0.w;
  a[4]  = (_Float16)f1.x; a[5]  = (_Float16)f1.y;
  a[6]  = (_Float16)f1.z; a[7]  = (_Float16)f1.w;
  a[8]  = (_Float16)f2.x; a[9]  = (_Float16)f2.y;
  a[10] = (_Float16)f2.z; a[11] = (_Float16)f2.w;
  a[12] = (_Float16)f3.x; a[13] = (_Float16)f3.y;
  a[14] = (_Float16)f3.z; a[15] = (_Float16)f3.w;
  return a;
}

__global__ __launch_bounds__(kWaves * 32, 1)
void tt_aggregate_kernel(const float* __restrict__ H,    // [N, 8, 256]
                         const float* __restrict__ Uf,   // [256, 8]
                         const float* __restrict__ Ul,   // [8, 256, 64]
                         const float* __restrict__ Ula,  // [8, 256]
                         float* __restrict__ out) {      // [N, 256]
  // B fragments in LDS: 32 lanes x 16 halves contiguous per lane per fragment.
  __shared__ __align__(32) _Float16 sU[7 * 32 * 512];    // U_list chain   229376 B
  __shared__ __align__(32) _Float16 sUf[8 * 512];        // U_first (padded)  8192 B
  __shared__ __align__(32) _Float16 sUla[16 * 512];      // U_last  (padded) 16384 B
  __shared__ float sM[kWaves][16 * 64];                  // per-wave M scratch
  __shared__ float sR[kWaves][16 * 8];                   // per-wave r vectors

  // ---- cooperative stage: U_list f32 -> f16 fragment layout ----
  for (int e = threadIdx.x; e < 7 * 16384; e += blockDim.x) {
    const int d7  = e >> 14;          // 0..6  -> U_list[d7+1]
    const int rem = e & 16383;
    const int k   = rem >> 6;         // 0..255
    const int c   = rem & 63;         // 0..63
    const float v = Ul[(d7 + 1) * 16384 + rem];
    const int jt = c >> 4, n = c & 15;
    const int kb = k >> 5, kl = k & 31;
    const int ln = n + ((kl >> 4) << 4);            // lane: n + 16*khalf
    const int frag = (d7 * 4 + jt) * 8 + kb;
    sU[frag * 512 + ln * 16 + (kl & 15)] = (_Float16)v;  // element = k&15
  }
  // ---- U_first fragments, columns zero-padded 8 -> 16 (loads hoisted; no divergence) ----
  for (int e = threadIdx.x; e < 8 * 512; e += blockDim.x) {
    const int kb = e >> 9, pos = e & 511;
    const int ln = pos >> 4, kk = pos & 15;
    const int n = ln & 15, khalf = ln >> 4;
    const int k = kb * 32 + khalf * 16 + kk;
    const float x = Uf[k * kR + (n & 7)];           // always-valid address
    sUf[e] = (_Float16)((n < kR) ? x : 0.0f);
  }
  // ---- U_last fragments, K zero-padded 8 -> 32 ----
  for (int e = threadIdx.x; e < 16 * 512; e += blockDim.x) {
    const int ot = e >> 9, pos = e & 511;
    const int ln = pos >> 4, kk = pos & 15;
    const int n = ln & 15, khalf = ln >> 4;
    const int k = khalf * 16 + kk;
    const float x = Ula[(k & 7) * kOut + ot * 16 + n];  // always-valid address
    sUla[e] = (_Float16)((k < kR) ? x : 0.0f);
  }
  __syncthreads();

  const int lane = threadIdx.x & 31;
  const int wave = threadIdx.x >> 5;
  const int tile = blockIdx.x * kWaves + wave;
  if (tile >= kTiles) return;        // wave-uniform; no barriers past this point

  const int m = lane & 15;           // sample-in-tile (A rows / C columns)
  const int g = lane >> 4;           // lane half (K-group select)
  int row = tile * 16 + m;
  if (row >= kN) row = kN - 1;
  const float* hrow = H + (long)row * (kD * kIn);
  float* sMw = &sM[wave][0];
  float* sRw = &sR[wave][0];
  const _Float16* bl = &sU[lane * 16];

  __builtin_prefetch(hrow, 0, 0);
  __builtin_prefetch(hrow + kIn, 0, 0);

  // ---- stage 1: r0 = h0 @ U_first ----
  {
    v8f acc = {};
#pragma unroll
    for (int kb = 0; kb < 8; ++kb) {
      const v16h a = make_a_frag(hrow, kb, g);
      const v16h b = *(const v16h*)&sUf[kb * 512 + lane * 16];
      WMMA_F32_F16(acc, a, b);
    }
    if (m < kR) {
#pragma unroll
      for (int v = 0; v < 8; ++v) sRw[(v + g * 8) * kR + m] = acc[v];
    }
  }

  // ---- stage 2: chain d = 1..7: M = h_d @ U_list[d]; r = M @ r ----
  for (int d = 1; d < kD; ++d) {
    const float* hp = hrow + d * kIn;
    __builtin_prefetch(hrow + ((d + 1 < kD) ? (d + 1) : d) * kIn, 0, 0);

    v16h afr[8];
#pragma unroll
    for (int kb = 0; kb < 8; ++kb) afr[kb] = make_a_frag(hp, kb, g);

#pragma unroll
    for (int jt = 0; jt < 4; ++jt) {
      v8f acc = {};
      const _Float16* bbase = bl + (((d - 1) * 4 + jt) * 8) * 512;
#pragma unroll
      for (int kb = 0; kb < 8; ++kb) {
        const v16h b = *(const v16h*)(bbase + kb * 512);
        WMMA_F32_F16(acc, afr[kb], b);
      }
#pragma unroll
      for (int v = 0; v < 8; ++v)              // C: sample = v + 8*g, col = m
        sMw[(v + g * 8) * kRR + jt * 16 + m] = acc[v];
    }

    // r <- M r : 2 lanes per sample, 4 output components per lane (in-wave, DS in order)
    float rl[8];
#pragma unroll
    for (int j = 0; j < 8; ++j) rl[j] = sRw[m * kR + j];
    float rn[4];
#pragma unroll
    for (int ii = 0; ii < 4; ++ii) {
      const int i = g * 4 + ii;
      float t = 0.0f;
#pragma unroll
      for (int j = 0; j < 8; ++j) t += sMw[m * kRR + i * kR + j] * rl[j];
      rn[ii] = t;
    }
#pragma unroll
    for (int ii = 0; ii < 4; ++ii) sRw[m * kR + g * 4 + ii] = rn[ii];
  }

  // ---- stage 3: out = r @ U_last ----
  {
    float rv[8];
#pragma unroll
    for (int j = 0; j < 8; ++j) rv[j] = sRw[m * kR + j];
    v16h a;
#pragma unroll
    for (int e = 0; e < 16; ++e)               // K = g*16 + e; valid K<8
      a[e] = (_Float16)((g == 0 && e < 8) ? rv[e & 7] : 0.0f);

#pragma unroll
    for (int ot = 0; ot < 16; ++ot) {
      const v16h b = *(const v16h*)&sUla[ot * 512 + lane * 16];
      v8f acc = {};
      WMMA_F32_F16(acc, a, b);
#pragma unroll
      for (int v = 0; v < 8; ++v) {            // in-bounds: kN is a multiple of 16
        const long r = (long)tile * 16 + v + g * 8;
        out[r * kOut + ot * 16 + m] = acc[v];
      }
    }
  }
}

extern "C" void kernel_launch(void* const* d_in, const int* in_sizes, int n_in,
                              void* d_out, int out_size, void* d_ws, size_t ws_size,
                              hipStream_t stream) {
  const float* H   = (const float*)d_in[0];   // neighbour_states [N,8,256]
  const float* Uf  = (const float*)d_in[1];   // U_first [256,8]
  const float* Ul  = (const float*)d_in[2];   // U_list  [8,256,64]
  const float* Ula = (const float*)d_in[3];   // U_last  [8,256]
  float* out = (float*)d_out;                 // [N,256]
  (void)in_sizes; (void)n_in; (void)out_size; (void)d_ws; (void)ws_size;

  const int blocks = (kTiles + kWaves - 1) / kWaves;  // 782
  tt_aggregate_kernel<<<blocks, kWaves * 32, 0, stream>>>(H, Uf, Ul, Ula, out);
}
